// LatentSDE_36146444763917
// MI455X (gfx1250) — matrix operand order repaired and checked by reading, not compiled, not run on hardware
//
#include <hip/hip_runtime.h>
#include <math.h>

// Problem constants (from the reference)
#define Dd 32      // latent dim
#define Bb 4096    // batch
#define Tt 256     // time steps
#define Hh 64      // hidden

typedef __attribute__((ext_vector_type(16))) _Float16 v16h;
typedef __attribute__((ext_vector_type(8)))  _Float16 v8h;
typedef __attribute__((ext_vector_type(8)))  float    v8f;

#if __has_builtin(__builtin_amdgcn_tanhf)
#define TANHF(x) __builtin_amdgcn_tanhf(x)
#else
#define TANHF(x) tanhf(x)
#endif

// Build a 16-element f16 A/B fragment from two contiguous 16B LDS chunks.
__device__ __forceinline__ v16h ldfrag(const _Float16* p0, const _Float16* p1) {
  v8h a = *(const v8h*)p0;
  v8h b = *(const v8h*)p1;
  return __builtin_shufflevector(a, b, 0,1,2,3,4,5,6,7,8,9,10,11,12,13,14,15);
}

__device__ __forceinline__ v8f wmma16(v16h A, v16h B, v8f C) {
  // D = A(16x32 f16) * B(32x16 f16) + C(16x16 f32)
  return __builtin_amdgcn_wmma_f32_16x16x32_f16(false, A, false, B, (short)0, C,
                                                false, false);
}

__global__ __launch_bounds__(32)
void latent_sde_kernel(const float* __restrict__ z0, const float* __restrict__ ts,
                       const float* __restrict__ noise,
                       const float* __restrict__ W1, const float* __restrict__ b1,
                       const float* __restrict__ W2, const float* __restrict__ b2,
                       const float* __restrict__ W3, const float* __restrict__ b3,
                       const float* __restrict__ log_std,
                       float* __restrict__ out)
{
  // Transposed f16 weights, staged once, reused 255 steps.
  __shared__ __align__(16) _Float16 sW1T[Hh * Dd];   // [c][k] : W1[1+k][c]
  __shared__ __align__(16) _Float16 sW2T[Hh * Hh];   // [c][k] : W2[k][c]
  __shared__ __align__(16) _Float16 sW3T[Dd * Hh];   // [c][k] : W3[k][c]
  // Per-wave activation tiles (row-major [m][k], f16).
  __shared__ __align__(16) _Float16 sY [16 * Dd];
  __shared__ __align__(16) _Float16 sH1[16 * Hh];
  __shared__ __align__(16) _Float16 sH2[16 * Hh];

  const int lane = threadIdx.x;       // wave32: 0..31
  const int half = lane >> 4;         // 0 / 1 (lane group)
  const int col  = lane & 15;         // N / M index within group
  const int rowbase = blockIdx.x * 16;

  // ---- stage weights (transposed, f16) into LDS ----
  for (int i = lane; i < Hh*Dd; i += 32) { int c = i / Dd, k = i % Dd; sW1T[i] = (_Float16)W1[(k + 1)*Hh + c]; }
  for (int i = lane; i < Hh*Hh; i += 32) { int c = i >> 6, k = i & 63;  sW2T[i] = (_Float16)W2[k*Hh + c]; }
  for (int i = lane; i < Dd*Hh; i += 32) { int c = i >> 6, k = i & 63;  sW3T[i] = (_Float16)W3[k*Dd + c]; }

  // ---- per-lane bias / scale constants (indexed by output column) ----
  float b1c[4], w1tc[4], b2c[4], b3c[2], stdc[2];
  #pragma unroll
  for (int nt = 0; nt < 4; ++nt) { int c = col + 16*nt; b1c[nt] = b1[c]; w1tc[nt] = W1[c]; b2c[nt] = b2[c]; }
  #pragma unroll
  for (int nt = 0; nt < 2; ++nt) { int c = col + 16*nt; b3c[nt] = b3[c]; stdc[nt] = expf(log_std[c]); }

  // ---- y state in fp32, kept in the WMMA C/D register layout ----
  // y[nt][i] holds element (M = i + 8*half, N = col + 16*nt) of the 16x32 tile.
  float y[2][8];
  #pragma unroll
  for (int nt = 0; nt < 2; ++nt)
    #pragma unroll
    for (int i = 0; i < 8; ++i) {
      int m = i + 8*half;
      size_t g = (size_t)(rowbase + m) * Dd + col + 16*nt;
      float v = z0[g];
      y[nt][i] = v;
      out[g] = v;                              // ys[0] = z0
      sY[m*Dd + col + 16*nt] = (_Float16)v;    // f16 copy for the GEMM
    }
  __syncthreads();   // single-wave WG: compiler fence, hw S_NOP

  for (int s = 0; s < Tt - 1; ++s) {
    float t  = ts[s];
    float dt = ts[s + 1] - t;
    float sq = sqrtf(dt);

    // Load this step's noise early; it is consumed only after layer 3,
    // so its latency hides under the GEMM chain.
    const float* np = noise + (size_t)s * Bb * Dd;
    float nz[2][8];
    #pragma unroll
    for (int nt = 0; nt < 2; ++nt)
      #pragma unroll
      for (int i = 0; i < 8; ++i) {
        int m = i + 8*half;
        nz[nt][i] = np[(size_t)(rowbase + m)*Dd + col + 16*nt];
      }
    if (s + 1 < Tt - 1)  // prefetch next step's noise tile (uniform branch)
      __builtin_prefetch(np + (size_t)Bb*Dd + (size_t)(rowbase + col)*Dd, 0, 0);

    // ---- layer 1: h1 = tanh(y @ W1[1:,:] + (b1 + t*W1[0,:])), K = 32 ----
    v16h Ay = ldfrag(sY + col*Dd + 8*half, sY + col*Dd + 8*half + 16);
    #pragma unroll
    for (int nt = 0; nt < 4; ++nt) {
      int c = col + 16*nt;
      v16h Bf = ldfrag(sW1T + c*Dd + 8*half, sW1T + c*Dd + 8*half + 16);
      v8f acc = {};
      acc = wmma16(Ay, Bf, acc);
      float bias = b1c[nt] + t * w1tc[nt];
      #pragma unroll
      for (int i = 0; i < 8; ++i) {
        int m = i + 8*half;
        sH1[m*Hh + c] = (_Float16)TANHF(acc[i] + bias);
      }
    }
    __syncthreads();

    // ---- layer 2: h2 = tanh(h1 @ W2 + b2), K = 64 (two K-steps) ----
    v16h A20 = ldfrag(sH1 + col*Hh + 8*half,      sH1 + col*Hh + 8*half + 16);
    v16h A21 = ldfrag(sH1 + col*Hh + 32 + 8*half, sH1 + col*Hh + 32 + 8*half + 16);
    #pragma unroll
    for (int nt = 0; nt < 4; ++nt) {
      int c = col + 16*nt;
      v8f acc = {};
      v16h B0 = ldfrag(sW2T + c*Hh + 8*half,      sW2T + c*Hh + 8*half + 16);
      acc = wmma16(A20, B0, acc);
      v16h B1 = ldfrag(sW2T + c*Hh + 32 + 8*half, sW2T + c*Hh + 32 + 8*half + 16);
      acc = wmma16(A21, B1, acc);
      #pragma unroll
      for (int i = 0; i < 8; ++i) {
        int m = i + 8*half;
        sH2[m*Hh + c] = (_Float16)TANHF(acc[i] + b2c[nt]);
      }
    }
    __syncthreads();

    // ---- layer 3 + Euler–Maruyama update (fp32 state) ----
    v16h A30 = ldfrag(sH2 + col*Hh + 8*half,      sH2 + col*Hh + 8*half + 16);
    v16h A31 = ldfrag(sH2 + col*Hh + 32 + 8*half, sH2 + col*Hh + 32 + 8*half + 16);
    float* os = out + (size_t)(s + 1) * Bb * Dd;
    #pragma unroll
    for (int nt = 0; nt < 2; ++nt) {
      int c = col + 16*nt;
      v8f acc = {};
      v16h B0 = ldfrag(sW3T + c*Hh + 8*half,      sW3T + c*Hh + 8*half + 16);
      acc = wmma16(A30, B0, acc);
      v16h B1 = ldfrag(sW3T + c*Hh + 32 + 8*half, sW3T + c*Hh + 32 + 8*half + 16);
      acc = wmma16(A31, B1, acc);
      #pragma unroll
      for (int i = 0; i < 8; ++i) {
        int m = i + 8*half;
        float drift = acc[i] + b3c[nt];
        float yn = y[nt][i] + drift*dt + stdc[nt]*sq*nz[nt][i];
        y[nt][i] = yn;
        os[(size_t)(rowbase + m)*Dd + c] = yn;     // ys[s+1]
        sY[m*Dd + c] = (_Float16)yn;               // f16 input for next step
      }
    }
    __syncthreads();
  }
}

extern "C" void kernel_launch(void* const* d_in, const int* in_sizes, int n_in,
                              void* d_out, int out_size, void* d_ws, size_t ws_size,
                              hipStream_t stream) {
  const float* z0    = (const float*)d_in[0];
  const float* ts    = (const float*)d_in[1];
  const float* noise = (const float*)d_in[2];
  const float* W1    = (const float*)d_in[3];
  const float* b1    = (const float*)d_in[4];
  const float* W2    = (const float*)d_in[5];
  const float* b2    = (const float*)d_in[6];
  const float* W3    = (const float*)d_in[7];
  const float* b3    = (const float*)d_in[8];
  const float* lstd  = (const float*)d_in[9];
  float* out = (float*)d_out;

  dim3 grid(Bb / 16);   // 256 independent single-wave workgroups
  dim3 block(32);       // wave32
  hipLaunchKernelGGL(latent_sde_kernel, grid, block, 0, stream,
                     z0, ts, noise, W1, b1, W2, b2, W3, b3, lstd, out);
}
